// GCN_44306882626192
// MI455X (gfx1250) — matrix operand order repaired and checked by reading, not compiled, not run on hardware
//
#include <hip/hip_runtime.h>

#define NN 16384
#define DD 128
#define BM 64      // adjacency rows per block
#define KT 128     // K tile (reduction dim per stage)

typedef __attribute__((ext_vector_type(16))) __bf16 bf16x16;
typedef __attribute__((ext_vector_type(8)))  __bf16 bf16x8;
typedef __attribute__((ext_vector_type(8)))  float  f32x8;
typedef __attribute__((ext_vector_type(4)))  float  f32x4;

union Frag { bf16x16 v; bf16x8 h[2]; };

__device__ __forceinline__ unsigned short f2bf(float f) {
  union { float f; unsigned u; } c; c.f = f;
  unsigned u = c.u;
  return (unsigned short)((u + 0x7FFFu + ((u >> 16) & 1u)) >> 16);  // RNE
}
__device__ __forceinline__ unsigned pk2bf(float a, float b) {
  return (unsigned)f2bf(a) | ((unsigned)f2bf(b) << 16);
}

// 8 x b128 async global->LDS copies (128 bytes), ASYNCcnt-tracked.
__device__ __forceinline__ void async_copy_x(unsigned ldsOff, const unsigned short* gp) {
  unsigned long long ga = (unsigned long long)(uintptr_t)gp;
  asm volatile(
      "global_load_async_to_lds_b128 %0, %1, off\n\t"
      "global_load_async_to_lds_b128 %0, %1, off offset:16\n\t"
      "global_load_async_to_lds_b128 %0, %1, off offset:32\n\t"
      "global_load_async_to_lds_b128 %0, %1, off offset:48\n\t"
      "global_load_async_to_lds_b128 %0, %1, off offset:64\n\t"
      "global_load_async_to_lds_b128 %0, %1, off offset:80\n\t"
      "global_load_async_to_lds_b128 %0, %1, off offset:96\n\t"
      "global_load_async_to_lds_b128 %0, %1, off offset:112\n\t"
      :: "v"(ldsOff), "v"(ga) : "memory");
}
__device__ __forceinline__ void wait_async0() {
  asm volatile("s_wait_asynccnt 0x0" ::: "memory");
}

// A fragment: 16x32 bf16, lane holds K[lhi*8..+8) and K[16+lhi*8..+8) of row.
__device__ __forceinline__ void loadA(const unsigned short* base, int stride,
                                      int row, int ks, int lhi, Frag& f) {
  const bf16x8* p = (const bf16x8*)(base + row * stride + ks + lhi * 8);
  f.h[0] = p[0];
  f.h[1] = p[2];
}
// B fragment: 32x16 bf16, lane holds K[lhi*8..+8) and K[16+lhi*8..+8) of column.
__device__ __forceinline__ void loadB(const unsigned short* base, int stride,
                                      int col, int ks, int lhi, Frag& f) {
  const bf16x8* p = (const bf16x8*)(base + col * stride + ks + lhi * 8);
  f.h[0] = p[0];
  f.h[1] = p[2];
}

// dst[c*R + r] = bf16(src[r*C + c]); grid = (C/32, R/32), block = (32,8)
__global__ void transpose_f32_to_bf16(const float* __restrict__ src,
                                      unsigned short* __restrict__ dst,
                                      int R, int C) {
  __shared__ unsigned short tile[32][33];
  int c0 = blockIdx.x * 32, r0 = blockIdx.y * 32;
  int tx = threadIdx.x, ty = threadIdx.y;
#pragma unroll
  for (int j = 0; j < 4; ++j) {
    int r = r0 + ty + j * 8;
    tile[ty + j * 8][tx] = f2bf(src[(size_t)r * C + c0 + tx]);
  }
  __syncthreads();
#pragma unroll
  for (int j = 0; j < 4; ++j) {
    int c = c0 + ty + j * 8;
    dst[(size_t)c * R + r0 + tx] = tile[tx][ty + j * 8];
  }
}

// Fused: out = relu(adj @ x @ W + b)
// Block: BM=64 adj rows, 8 waves; wave -> 16 rows x 64 cols (4 WMMA tiles).
__global__ __launch_bounds__(256)
void gcn_fused(const float* __restrict__ adj,
               const unsigned short* __restrict__ xT,   // bf16 [DD][NN]
               const unsigned short* __restrict__ wT,   // bf16 [UNITS][DD]
               const float* __restrict__ bias,
               float* __restrict__ out) {
  __shared__ union {
    struct {
      unsigned short a[2][BM * KT];   // adj bf16, 2 x 16 KB
      unsigned short b[2][DD * KT];   // x^T bf16, 2 x 32 KB
    } st;
    unsigned short agg[BM * DD];      // 16 KB overlay for fused GEMM2
  } sh;
  __shared__ unsigned short sW[DD * DD];  // W^T bf16, 32 KB

  const int t    = threadIdx.x;
  const int lane = t & 31;
  const int wv   = t >> 5;
  const int llo  = lane & 15;
  const int lhi  = lane >> 4;
  const int mrow = (wv >> 1) * 16;   // wave's row strip within block
  const int nh   = (wv & 1) * 64;    // wave's column half
  const size_t rowBase = (size_t)blockIdx.x * BM;

  // Stage W^T into LDS once
  {
    const uint4* s = (const uint4*)wT;
    uint4*       d = (uint4*)sW;
#pragma unroll
    for (int i = 0; i < 8; ++i) d[t * 8 + i] = s[t * 8 + i];
  }

  const f32x8 zero = {0.f, 0.f, 0.f, 0.f, 0.f, 0.f, 0.f, 0.f};
  f32x8 acc[4];
#pragma unroll
  for (int i = 0; i < 4; ++i) acc[i] = zero;

  // Per-thread staging assignments
  const int arow = t >> 2;            // 0..63   (adj tile row)
  const int acs  = (t & 3) * 32;      // 0/32/64/96 (adj col segment, 32 floats)
  const float* aptr = adj + (rowBase + arow) * (size_t)NN + acs;
  const int xn  = t >> 1;             // 0..127  (x^T feature row)
  const int xks = (t & 1) * 64;       // 0/64    (k segment, 64 bf16 = 128 B)
  const unsigned short* xp = xT + (size_t)xn * NN + xks;
  const unsigned ldsB[2] = {
      (unsigned)(uintptr_t)&sh.st.b[0][xn * KT + xks],
      (unsigned)(uintptr_t)&sh.st.b[1][xn * KT + xks]};

  // Prologue: tile 0 in flight (adjacency stream is single-use -> NT loads)
  f32x4 areg[8];
#pragma unroll
  for (int i = 0; i < 8; ++i)
    areg[i] = __builtin_nontemporal_load(((const f32x4*)aptr) + i);
  async_copy_x(ldsB[0], xp);

  for (int it = 0; it < NN / KT; ++it) {
    const int buf = it & 1;

    // convert + store adjacency tile (waits the global loads via loadcnt)
    {
      unsigned* d = (unsigned*)sh.st.a[buf] + arow * (KT / 2) + (acs >> 1);
#pragma unroll
      for (int i = 0; i < 8; ++i) {
        f32x4 v = areg[i];
        d[i * 2 + 0] = pk2bf(v.x, v.y);
        d[i * 2 + 1] = pk2bf(v.z, v.w);
      }
    }
    wait_async0();          // x^T copy for this buffer complete
    __syncthreads();

    // issue next tile (overlaps with compute below)
    if (it + 1 < NN / KT) {
      const f32x4* s = (const f32x4*)(aptr + (it + 1) * KT);
#pragma unroll
      for (int i = 0; i < 8; ++i) areg[i] = __builtin_nontemporal_load(s + i);
      async_copy_x(ldsB[buf ^ 1], xp + (it + 1) * KT);
    }

    // GEMM1 MACs: 16 software-pipelined (ks,nt) steps, fragments double-buffered
    {
      const unsigned short* As = sh.st.a[buf];
      const unsigned short* Bs = sh.st.b[buf];
      Frag a[2], b[2];
      loadA(As, KT, mrow + llo, 0, lhi, a[0]);
      loadB(Bs, KT, nh + llo, 0, lhi, b[0]);
#pragma unroll
      for (int s = 0; s < 16; ++s) {
        const int nt = s & 3;
        if (s + 1 < 16) {
          const int nt1 = (s + 1) & 3;
          const int ks1 = ((s + 1) >> 2) * 32;
          if (nt1 == 0) loadA(As, KT, mrow + llo, ks1, lhi, a[((s + 1) >> 2) & 1]);
          loadB(Bs, KT, nh + nt1 * 16 + llo, ks1, lhi, b[(s + 1) & 1]);
        }
        acc[nt] = __builtin_amdgcn_wmma_f32_16x16x32_bf16(
            false, a[(s >> 2) & 1].v, false, b[s & 1].v, (short)0, acc[nt],
            false, false);
      }
    }
  }
  __syncthreads();

  // Spill agg (bf16) into LDS (overlays dead adj buffers)
#pragma unroll
  for (int nt = 0; nt < 4; ++nt) {
    int n = nh + nt * 16 + llo;
#pragma unroll
    for (int j = 0; j < 8; ++j) {
      int m = mrow + j + lhi * 8;   // C-layout: VGPR j, lane-half lhi
      sh.agg[m * DD + n] = f2bf(acc[nt][j]);
    }
  }
  __syncthreads();

  // GEMM2: relu(agg @ W + b), same pipelined fragment pattern
  f32x8 acc2[4];
#pragma unroll
  for (int i = 0; i < 4; ++i) acc2[i] = zero;
  {
    Frag a[2], b[2];
    loadA(sh.agg, DD, mrow + llo, 0, lhi, a[0]);
    loadB(sW, DD, nh + llo, 0, lhi, b[0]);
#pragma unroll
    for (int s = 0; s < 16; ++s) {
      const int nt = s & 3;
      if (s + 1 < 16) {
        const int nt1 = (s + 1) & 3;
        const int ks1 = ((s + 1) >> 2) * 32;
        if (nt1 == 0) loadA(sh.agg, DD, mrow + llo, ks1, lhi, a[((s + 1) >> 2) & 1]);
        loadB(sW, DD, nh + nt1 * 16 + llo, ks1, lhi, b[(s + 1) & 1]);
      }
      acc2[nt] = __builtin_amdgcn_wmma_f32_16x16x32_bf16(
          false, a[(s >> 2) & 1].v, false, b[s & 1].v, (short)0, acc2[nt],
          false, false);
    }
  }

  // bias + relu + store (f32, streaming -> NT stores)
#pragma unroll
  for (int nt = 0; nt < 4; ++nt) {
    int n = nh + nt * 16 + llo;
    float bv = bias[n];
#pragma unroll
    for (int j = 0; j < 8; ++j) {
      size_t m = rowBase + mrow + j + lhi * 8;
      float v = acc2[nt][j] + bv;
      __builtin_nontemporal_store(v > 0.f ? v : 0.f, &out[m * DD + n]);
    }
  }
}

extern "C" void kernel_launch(void* const* d_in, const int* in_sizes, int n_in,
                              void* d_out, int out_size, void* d_ws, size_t ws_size,
                              hipStream_t stream) {
  (void)in_sizes; (void)n_in; (void)out_size; (void)ws_size;
  const float* x    = (const float*)d_in[0];   // [NN, DD]
  const float* adj  = (const float*)d_in[1];   // [NN, NN]
  const float* kern = (const float*)d_in[2];   // [DD, UNITS]
  const float* bias = (const float*)d_in[3];   // [UNITS]
  float* out = (float*)d_out;                  // [NN, UNITS]

  unsigned short* wT = (unsigned short*)d_ws;  // bf16 [UNITS][DD]   (32 KB)
  unsigned short* xT = wT + DD * DD;           // bf16 [DD][NN]      (4 MB)

  transpose_f32_to_bf16<<<dim3(DD / 32, NN / 32), dim3(32, 8), 0, stream>>>(x, xT, NN, DD);
  transpose_f32_to_bf16<<<dim3(DD / 32, DD / 32), dim3(32, 8), 0, stream>>>(kern, wT, DD, DD);
  gcn_fused<<<NN / BM, 256, 0, stream>>>(adj, xT, wT, bias, out);
}
